// SpatioTemporalModel_27848567947367
// MI455X (gfx1250) — matrix operand order repaired and checked by reading, compiled, not run on hardware
//
#include <hip/hip_runtime.h>
#include <hip/hip_bf16.h>

typedef __attribute__((ext_vector_type(16))) _Float16 v16h;
typedef __attribute__((ext_vector_type(8)))  _Float16 v8h;
typedef __attribute__((ext_vector_type(8)))  float    v8f;
typedef __attribute__((ext_vector_type(4)))  int      v4i;

#define DEV static __device__ __forceinline__

// ------------------------------------------------- async global->LDS support
#if defined(__has_builtin)
#  if __has_builtin(__builtin_amdgcn_global_load_async_to_lds_b128)
#    define HAS_ASYNC_COPY 1
#  endif
#endif

#if HAS_ASYNC_COPY
typedef __attribute__((address_space(1))) v4i gas_v4i;  // global int4
typedef __attribute__((address_space(3))) v4i las_v4i;  // LDS int4

DEV void async_copy_b128(const float* g, float* l) {
  __builtin_amdgcn_global_load_async_to_lds_b128(
      (gas_v4i*)(void*)g, (las_v4i*)(void*)l, 0, 0);
}
#  if __has_builtin(__builtin_amdgcn_s_wait_asynccnt)
#    define WAIT_ASYNC0() __builtin_amdgcn_s_wait_asynccnt(0)
#  else
#    define WAIT_ASYNC0() asm volatile("s_wait_asynccnt 0x0" ::: "memory")
#  endif
#endif

// ---------------------------------------------------------------- WMMA helpers
DEV v8f wmma16(v16h a, v16h b, v8f c) {
  return __builtin_amdgcn_wmma_f32_16x16x32_f16(false, a, false, b, (short)0, c,
                                                false, false);
}

// A fragment (16x32 f16) from an LDS row-major tile.
// Layout (ISA 7.12.2): lane&15 = row; lanes 0-15 hold K 0-7 (v0-3) & 16-23 (v4-7),
// lanes 16-31 hold K 8-15 & 24-31.  stride/kbase are 16B multiples -> v8h loads.
DEV v16h ldsA(const _Float16* base, int stride, int mbase, int kbase, int lane) {
  int row  = mbase + (lane & 15);
  int kofs = (lane >> 4) << 3;
  const _Float16* p = base + row * stride + kbase + kofs;
  v8h lo = *(const v8h*)p;
  v8h hi = *(const v8h*)(p + 16);
  v16h a;
#pragma unroll
  for (int e = 0; e < 8; ++e) { a[e] = lo[e]; a[8 + e] = hi[e]; }
  return a;
}

// Packed B fragment: fragment f stored as [32 lanes][16 halves] contiguously.
DEV v16h ldB(const _Float16* pk, int frag, int lane) {
  return *(const v16h*)(pk + (size_t)frag * 512 + lane * 16);
}

// ---------------------------------------------------------------- constants
#define NN     2048
#define LL     100
#define MROWS  204800      // N*L
#define MAXNBR 96

// workspace byte offsets (all 256-aligned)
#define O_B    ((size_t)0)                    // B-state / Y (in-place) [M,16] f32
#define O_C    (O_B   + 13107200)             // C-state [M,16] f32
#define O_XT1  (O_C   + 13107200)             // xt1 [4,2048,64] f32
#define O_H1   (O_XT1 + 2097152)              // h1  [2048,256]  f32
#define O_XT2  (O_H1  + 2097152)              // xt2 [2048,64]   f32
#define O_WEF  (O_XT2 + 524288)               // Weff [16,128] f32
#define O_BEF  (O_WEF + 8192)                 // beff [128] f32
#define O_E1S  (O_BEF + 512)
#define O_E1D  (O_E1S + 32768)
#define O_E2S  (O_E1D + 32768)
#define O_E2D  (O_E2S + 8192)
#define O_NBR  (O_E2D + 8192)                 // [2048,96] int
#define O_CNT  (O_NBR + 786432)               // [2048] int
#define O_PWI  (O_CNT + 8192)                 // packed f16 weights below
#define O_PWG  (O_PWI + 65536)
#define O_PWB  (O_PWG + 65536)
#define O_PWC  (O_PWB + 8192)
#define O_PWE  (O_PWC + 4096)
#define O_PW1  (O_PWE + 8192)
#define O_PW2  (O_PW1 + 65536)

// ---------------------------------------------------------------- prep kernels
__global__ __launch_bounds__(256) void k_build_weff(
    const float* __restrict__ Wo, const float* __restrict__ Wfc,
    const float* __restrict__ bo, const float* __restrict__ bfc,
    float* __restrict__ Weff, float* __restrict__ beff) {
  int idx = blockIdx.x * 256 + threadIdx.x;
  if (idx < 2048) {
    int s = idx >> 7, o = idx & 127;
    float acc = 0.f;
    for (int j = 0; j < 128; ++j) acc += Wo[s * 128 + j] * Wfc[j * 128 + o];
    Weff[idx] = acc;
  } else if (idx < 2176) {
    int o = idx - 2048;
    float acc = bfc[o];
    for (int j = 0; j < 128; ++j) acc += bo[j] * Wfc[j * 128 + o];
    beff[o] = acc;
  }
}

// Pack an f32 [K,N] (row-major) weight into per-lane WMMA B-fragments (f16),
// K zero-padded to a multiple of 32.
__global__ __launch_bounds__(256) void k_pack(const float* __restrict__ W,
                                              int K, int N,
                                              _Float16* __restrict__ dst) {
  int Kp = (K + 31) & ~31;
  int total = Kp * N;
  int idx = blockIdx.x * 256 + threadIdx.x;
  if (idx >= total) return;
  int e = idx & 15, lane = (idx >> 4) & 31, frag = idx >> 9;
  int ntiles = N >> 4;
  int nt = frag % ntiles, kc = frag / ntiles;
  int k   = kc * 32 + ((lane >> 4) << 3) + (e & 7) + ((e >> 3) << 4);
  int col = nt * 16 + (lane & 15);
  dst[idx] = (_Float16)((k < K) ? W[(size_t)k * N + col] : 0.f);
}

// ---------------------------------------------------------------- mamba fused
// 32 rows/block, 8 waves. Computes x=t+pos, Xg=(xWi+bi)*sig(xWg+bg) (kept in
// LDS only), B=Xg*Wb+bb (LDS atomic reduce), C=x*Wc+bc.
__global__ __launch_bounds__(256) void k_mamba_gemm(
    const float* __restrict__ X, const float* __restrict__ pos,
    const float* __restrict__ bi, const float* __restrict__ bg,
    const float* __restrict__ bb, const float* __restrict__ bc,
    const _Float16* __restrict__ WiPk, const _Float16* __restrict__ WgPk,
    const _Float16* __restrict__ WbPk, const _Float16* __restrict__ WcPk,
    float* __restrict__ Bst, float* __restrict__ Cst) {
  __shared__ __align__(16) _Float16 sX[32 * 136];
  __shared__ __align__(16) _Float16 sXg[32 * 264];
  __shared__ float sB[32 * 16];
  int tid = threadIdx.x, blk = blockIdx.x;

  // phase 0: stage x = temporal + pos_emb as f16
#if HAS_ASYNC_COPY
  __shared__ __align__(16) float sXf[32 * 128];
  {
    const float* g = X + (size_t)blk * 4096 + tid * 4;
    float* l = sXf + tid * 4;
#pragma unroll
    for (int i = 0; i < 4; ++i) async_copy_b128(g + i * 1024, l + i * 1024);
    WAIT_ASYNC0();
  }
  __syncthreads();
#pragma unroll
  for (int e = tid; e < 4096; e += 256) {
    int r = e >> 7, c = e & 127;
    int rg = blk * 32 + r;
    sX[r * 136 + c] = (_Float16)(sXf[e] + pos[(rg % LL) * 128 + c]);
  }
#else
#pragma unroll
  for (int e = tid; e < 4096; e += 256) {
    int r = e >> 7, c = e & 127;
    int rg = blk * 32 + r;
    float v = X[(size_t)rg * 128 + c] + pos[(rg % LL) * 128 + c];
    sX[r * 136 + c] = (_Float16)v;
  }
#endif
  sB[tid] = 0.f;
  sB[tid + 256] = 0.f;
  __syncthreads();

  int w = tid >> 5, lane = tid & 31;
  int col = lane & 15, rbase = (lane >> 4) << 3;

  v8f zero = {};
  v8f ai[2][2] = {{zero, zero}, {zero, zero}};
  v8f ag[2][2] = {{zero, zero}, {zero, zero}};
#pragma unroll
  for (int kc = 0; kc < 4; ++kc) {
    v16h a0 = ldsA(sX, 136, 0, kc * 32, lane);
    v16h a1 = ldsA(sX, 136, 16, kc * 32, lane);
#pragma unroll
    for (int s = 0; s < 2; ++s) {
      v16h bif = ldB(WiPk, kc * 16 + w * 2 + s, lane);
      v16h bgf = ldB(WgPk, kc * 16 + w * 2 + s, lane);
      ai[0][s] = wmma16(a0, bif, ai[0][s]);
      ai[1][s] = wmma16(a1, bif, ai[1][s]);
      ag[0][s] = wmma16(a0, bgf, ag[0][s]);
      ag[1][s] = wmma16(a1, bgf, ag[1][s]);
    }
  }
  // gating epilogue -> sXg (f16), this wave's 32-col slice
#pragma unroll
  for (int m = 0; m < 2; ++m)
#pragma unroll
    for (int s = 0; s < 2; ++s) {
      int n = w * 32 + s * 16 + col;
      float bIv = bi[n], bGv = bg[n];
#pragma unroll
      for (int v = 0; v < 8; ++v) {
        float xi = ai[m][s][v] + bIv;
        float xg = ag[m][s][v] + bGv;
        float g = xi * (1.f / (1.f + __expf(-xg)));
        sXg[(m * 16 + rbase + v) * 264 + n] = (_Float16)g;
      }
    }
  // B-state partials: this wave's K-slice of Xg @ Wb, reduce via LDS atomics
  {
    v16h bwb = ldB(WbPk, w, lane);  // Wb: 8 k-chunks of 32, 1 n-tile
#pragma unroll
    for (int m = 0; m < 2; ++m) {
      v16h a = ldsA(sXg, 264, m * 16, w * 32, lane);
      v8f c = {};
      c = wmma16(a, bwb, c);
#pragma unroll
      for (int v = 0; v < 8; ++v)
        atomicAdd(&sB[(m * 16 + rbase + v) * 16 + col], c[v]);
    }
  }
  __syncthreads();
#pragma unroll
  for (int e = tid; e < 512; e += 256) {
    int r = e >> 4, s2 = e & 15;
    Bst[(size_t)(blk * 32 + r) * 16 + s2] = sB[e] + bb[s2];
  }
  // C-state: x @ Wc (+bc), waves 0..1 handle the two 16-row tiles
  if (w < 2) {
    v8f c = {};
#pragma unroll
    for (int kc = 0; kc < 4; ++kc) {
      v16h a = ldsA(sX, 136, w * 16, kc * 32, lane);
      v16h b = ldB(WcPk, kc, lane);
      c = wmma16(a, b, c);
    }
    float bcv = bc[col];
#pragma unroll
    for (int v = 0; v < 8; ++v)
      Cst[(size_t)(blk * 32 + w * 16 + rbase + v) * 16 + col] = c[v] + bcv;
  }
}

// selective-scan recurrence; Y overwrites B in place
__global__ __launch_bounds__(256) void k_scan(const float* __restrict__ Alog,
                                              const float* __restrict__ Cst,
                                              float* __restrict__ BY) {
  int idx = blockIdx.x * 256 + threadIdx.x;  // 2048*16
  int n = idx >> 4, s = idx & 15;
  float Ac = __expf(0.01f * __expf(Alog[s]));
  size_t base = (size_t)n * LL * 16 + s;
  float h = 0.f;
  for (int l = 0; l < LL; ++l) {
    size_t p = base + (size_t)l * 16;
    h = Ac * h + BY[p];
    BY[p] = h * Cst[p];
  }
}

// mamba_features = Y[M,16] @ Weff[16(pad32),128] + beff
__global__ __launch_bounds__(256) void k_feat(const float* __restrict__ Y,
                                              const _Float16* __restrict__ WePk,
                                              const float* __restrict__ beff,
                                              float* __restrict__ out) {
  __shared__ __align__(16) _Float16 sY[64 * 48];
  int tid = threadIdx.x, blk = blockIdx.x;
#if HAS_ASYNC_COPY
  __shared__ __align__(16) float sYf[64 * 16];
  async_copy_b128(Y + (size_t)blk * 1024 + tid * 4, sYf + tid * 4);
  WAIT_ASYNC0();
  __syncthreads();
  for (int e = tid; e < 1024; e += 256) {
    int r = e >> 4, k = e & 15;
    sY[r * 48 + k] = (_Float16)sYf[e];
    sY[r * 48 + 16 + k] = (_Float16)0.f;  // zero K pad 16..31
  }
#else
  for (int e = tid; e < 1024; e += 256) {
    int r = e >> 4, k = e & 15;
    sY[r * 48 + k] = (_Float16)Y[(size_t)(blk * 64 + r) * 16 + k];
    sY[r * 48 + 16 + k] = (_Float16)0.f;  // zero K pad 16..31
  }
#endif
  __syncthreads();
  int w = tid >> 5, lane = tid & 31, col = lane & 15, rbase = (lane >> 4) << 3;
  v16h b = ldB(WePk, w, lane);  // 1 k-chunk, 8 n-tiles -> frag = w
  float bev = beff[w * 16 + col];
#pragma unroll
  for (int m = 0; m < 4; ++m) {
    v16h a = ldsA(sY, 48, m * 16, 0, lane);
    v8f c = {};
    c = wmma16(a, b, c);
#pragma unroll
    for (int v = 0; v < 8; ++v)
      out[(size_t)(blk * 64 + m * 16 + rbase + v) * 128 + w * 16 + col] =
          c[v] + bev;
  }
}

// ---------------------------------------------------------------- GAT kernels
__global__ __launch_bounds__(256) void k_adj(const float* __restrict__ adj,
                                             int* __restrict__ nbr,
                                             int* __restrict__ ncnt) {
  int i = blockIdx.x * 256 + threadIdx.x;
  if (i >= NN) return;
  const float* row = adj + (size_t)i * NN;
  int cnt = 0;
  for (int j = 0; j < NN; ++j) {
    if (row[j] > 0.f) {
      if (cnt < MAXNBR) nbr[i * MAXNBR + cnt] = j;
      ++cnt;
    }
  }
  ncnt[i] = cnt < MAXNBR ? cnt : MAXNBR;
}

// layer-1 transform: xt1[h,n,o] = graph_x @ W1[h]
__global__ __launch_bounds__(256) void k_gat1(const float* __restrict__ Xg,
                                              const _Float16* __restrict__ W1Pk,
                                              float* __restrict__ xt1) {
  __shared__ __align__(16) _Float16 sX[64 * 136];
  int tid = threadIdx.x, blk = blockIdx.x;
  for (int e = tid; e < 8192; e += 256) {
    int r = e >> 7, c = e & 127;
    sX[r * 136 + c] = (_Float16)Xg[(size_t)(blk * 64 + r) * 128 + c];
  }
  __syncthreads();
  int w = tid >> 5, lane = tid & 31, col = lane & 15, rbase = (lane >> 4) << 3;
  int h = w >> 1, np = w & 1;
  const _Float16* pk = W1Pk + h * 8192;
  v8f zero = {};
  v8f acc[4][2] = {{zero, zero}, {zero, zero}, {zero, zero}, {zero, zero}};
#pragma unroll
  for (int kc = 0; kc < 4; ++kc) {
    v16h a[4];
#pragma unroll
    for (int m = 0; m < 4; ++m) a[m] = ldsA(sX, 136, m * 16, kc * 32, lane);
#pragma unroll
    for (int s = 0; s < 2; ++s) {
      v16h b = ldB(pk, kc * 4 + np * 2 + s, lane);
#pragma unroll
      for (int m = 0; m < 4; ++m) acc[m][s] = wmma16(a[m], b, acc[m][s]);
    }
  }
#pragma unroll
  for (int m = 0; m < 4; ++m)
#pragma unroll
    for (int s = 0; s < 2; ++s) {
      int o = np * 32 + s * 16 + col;
#pragma unroll
      for (int v = 0; v < 8; ++v) {
        int row = blk * 64 + m * 16 + rbase + v;
        xt1[(size_t)h * (NN * 64) + (size_t)row * 64 + o] = acc[m][s][v];
      }
    }
}

// e_src/e_dst = xt . a[:Fout] / a[Fout:]
__global__ __launch_bounds__(256) void k_edge(const float* __restrict__ xt,
                                              const float* __restrict__ av,
                                              int H, float* __restrict__ es,
                                              float* __restrict__ ed) {
  int idx = blockIdx.x * 256 + threadIdx.x;
  if (idx >= H * NN) return;
  int h = idx >> 11, n = idx & (NN - 1);
  const float* xp = xt + (size_t)h * (NN * 64) + (size_t)n * 64;
  const float* ah = av + h * 128;
  float s = 0.f, d = 0.f;
  for (int o = 0; o < 64; ++o) {
    float v = xp[o];
    s += v * ah[o];
    d += v * ah[64 + o];
  }
  es[idx] = s;
  ed[idx] = d;
}

// sparse masked-softmax attention; one wave per (head,row); lanes cover 64 dims
__global__ __launch_bounds__(256) void k_att(
    const float* __restrict__ xt, const float* __restrict__ es,
    const float* __restrict__ ed, const int* __restrict__ nbr,
    const int* __restrict__ ncnt, float* __restrict__ out, int outStride,
    int doRelu, int nTasks) {
  int task = blockIdx.x * 8 + (threadIdx.x >> 5);
  if (task >= nTasks) return;
  int lane = threadIdx.x & 31;
  int h = task >> 11, i = task & (NN - 1);
  int cnt = ncnt[i];
  float esi = es[task];
  const float* edh = ed + (h << 11);
  const int* nb = nbr + i * MAXNBR;
  float mx = -1e30f;
  for (int k = 0; k < cnt; ++k) {
    float sv = esi + edh[nb[k]];
    sv = sv > 0.f ? sv : 0.2f * sv;
    mx = fmaxf(mx, sv);
  }
  float sum = 0.f, a0 = 0.f, a1 = 0.f;
  const float* xh = xt + (size_t)h * (NN * 64);
  for (int k = 0; k < cnt; ++k) {
    int j = nb[k];
    float sv = esi + edh[j];
    sv = sv > 0.f ? sv : 0.2f * sv;
    float wv = __expf(sv - mx);
    sum += wv;
    const float* xp = xh + (size_t)j * 64;
    a0 += wv * xp[lane];
    a1 += wv * xp[lane + 32];
  }
  float inv = 1.f / sum;
  float r0 = a0 * inv, r1 = a1 * inv;
  if (doRelu) { r0 = fmaxf(r0, 0.f); r1 = fmaxf(r1, 0.f); }
  out[(size_t)i * outStride + h * 64 + lane] = r0;
  out[(size_t)i * outStride + h * 64 + lane + 32] = r1;
}

// layer-2 transform: xt2 = h1[2048,256] @ W2[256,64]
__global__ __launch_bounds__(256) void k_gat2(const float* __restrict__ H1,
                                              const _Float16* __restrict__ W2Pk,
                                              float* __restrict__ xt2) {
  __shared__ __align__(16) _Float16 sH[64 * 264];
  int tid = threadIdx.x, blk = blockIdx.x;
  for (int e = tid; e < 16384; e += 256) {
    int r = e >> 8, c = e & 255;
    sH[r * 264 + c] = (_Float16)H1[(size_t)(blk * 64 + r) * 256 + c];
  }
  __syncthreads();
  int w = tid >> 5, lane = tid & 31, col = lane & 15, rbase = (lane >> 4) << 3;
  int nt = w & 3, mg = w >> 2;
  v8f zero = {};
  v8f acc[2] = {zero, zero};
#pragma unroll
  for (int kc = 0; kc < 8; ++kc) {
    v16h b = ldB(W2Pk, kc * 4 + nt, lane);
    v16h a0 = ldsA(sH, 264, (mg * 2) * 16, kc * 32, lane);
    v16h a1 = ldsA(sH, 264, (mg * 2 + 1) * 16, kc * 32, lane);
    acc[0] = wmma16(a0, b, acc[0]);
    acc[1] = wmma16(a1, b, acc[1]);
  }
#pragma unroll
  for (int m = 0; m < 2; ++m)
#pragma unroll
    for (int v = 0; v < 8; ++v) {
      int row = blk * 64 + (mg * 2 + m) * 16 + rbase + v;
      xt2[(size_t)row * 64 + nt * 16 + col] = acc[m][v];
    }
}

// ---------------------------------------------------------------- launcher
extern "C" void kernel_launch(void* const* d_in, const int* in_sizes, int n_in,
                              void* d_out, int out_size, void* d_ws,
                              size_t ws_size, hipStream_t stream) {
  (void)in_sizes; (void)n_in; (void)out_size; (void)ws_size;
  const float* temporal = (const float*)d_in[0];
  const float* graph_x  = (const float*)d_in[1];
  const float* adj      = (const float*)d_in[2];
  const float* pos      = (const float*)d_in[3];
  const float* Wi  = (const float*)d_in[4];
  const float* bi  = (const float*)d_in[5];
  const float* Wg  = (const float*)d_in[6];
  const float* bg  = (const float*)d_in[7];
  const float* Alg = (const float*)d_in[8];
  const float* Wb  = (const float*)d_in[9];
  const float* bb  = (const float*)d_in[10];
  const float* Wc  = (const float*)d_in[11];
  const float* bc  = (const float*)d_in[12];
  const float* Wo  = (const float*)d_in[13];
  const float* bo  = (const float*)d_in[14];
  const float* Wfc = (const float*)d_in[15];
  const float* bfc = (const float*)d_in[16];
  const float* W1  = (const float*)d_in[17];
  const float* a1  = (const float*)d_in[18];
  const float* W2  = (const float*)d_in[19];
  const float* a2  = (const float*)d_in[20];

  char* ws = (char*)d_ws;
  float* Bst   = (float*)(ws + O_B);
  float* Cst   = (float*)(ws + O_C);
  float* xt1   = (float*)(ws + O_XT1);
  float* h1    = (float*)(ws + O_H1);
  float* xt2   = (float*)(ws + O_XT2);
  float* WeffF = (float*)(ws + O_WEF);
  float* beffF = (float*)(ws + O_BEF);
  float* e1s   = (float*)(ws + O_E1S);
  float* e1d   = (float*)(ws + O_E1D);
  float* e2s   = (float*)(ws + O_E2S);
  float* e2d   = (float*)(ws + O_E2D);
  int*   nbr   = (int*)(ws + O_NBR);
  int*   ncnt  = (int*)(ws + O_CNT);
  _Float16* pWi = (_Float16*)(ws + O_PWI);
  _Float16* pWg = (_Float16*)(ws + O_PWG);
  _Float16* pWb = (_Float16*)(ws + O_PWB);
  _Float16* pWc = (_Float16*)(ws + O_PWC);
  _Float16* pWe = (_Float16*)(ws + O_PWE);
  _Float16* pW1 = (_Float16*)(ws + O_PW1);
  _Float16* pW2 = (_Float16*)(ws + O_PW2);

  float* outM = (float*)d_out;
  float* outG = outM + (size_t)NN * LL * 128;

  // --- weight prep (Weff = Wo[0:16]@Wfc fold, then fragment packing)
  k_build_weff<<<9, 256, 0, stream>>>(Wo, Wfc, bo, bfc, WeffF, beffF);
  k_pack<<<128, 256, 0, stream>>>(Wi, 128, 256, pWi);
  k_pack<<<128, 256, 0, stream>>>(Wg, 128, 256, pWg);
  k_pack<<<16, 256, 0, stream>>>(Wb, 256, 16, pWb);
  k_pack<<<8, 256, 0, stream>>>(Wc, 128, 16, pWc);
  k_pack<<<16, 256, 0, stream>>>(WeffF, 16, 128, pWe);  // zero-padded to K=32
  for (int h = 0; h < 4; ++h)
    k_pack<<<32, 256, 0, stream>>>(W1 + h * 8192, 128, 64, pW1 + h * 8192);
  k_pack<<<64, 256, 0, stream>>>(W2, 256, 64, pW2);

  // --- mamba pipeline
  k_mamba_gemm<<<MROWS / 32, 256, 0, stream>>>(temporal, pos, bi, bg, bb, bc,
                                               pWi, pWg, pWb, pWc, Bst, Cst);
  k_scan<<<NN * 16 / 256, 256, 0, stream>>>(Alg, Cst, Bst);
  k_feat<<<MROWS / 64, 256, 0, stream>>>(Bst, pWe, beffF, outM);

  // --- GAT pipeline (sparse attention via capped CSR)
  k_adj<<<NN / 256, 256, 0, stream>>>(adj, nbr, ncnt);
  k_gat1<<<NN / 64, 256, 0, stream>>>(graph_x, pW1, xt1);
  k_edge<<<4 * NN / 256, 256, 0, stream>>>(xt1, a1, 4, e1s, e1d);
  k_att<<<4 * NN / 8, 256, 0, stream>>>(xt1, e1s, e1d, nbr, ncnt, h1, 256, 1,
                                        4 * NN);
  k_gat2<<<NN / 64, 256, 0, stream>>>(h1, pW2, xt2);
  k_edge<<<NN / 256, 256, 0, stream>>>(xt2, a2, 1, e2s, e2d);
  k_att<<<NN / 8, 256, 0, stream>>>(xt2, e2s, e2d, nbr, ncnt, outG, 64, 0, NN);
}
